// MyopicAttention_72060961292757
// MI455X (gfx1250) — matrix operand order repaired and checked by reading, not compiled, not run on hardware
//
#include <hip/hip_runtime.h>
#include <hip/hip_bf16.h>
#include <math.h>

// ---------------------------------------------------------------------------
// MyopicAttention for MI455X (gfx1250, wave32, WMMA bf16 16x16x32, f32 accum)
// B=4, N=2048, C=512, H=8, D=64, W=64, K=256, NW=32
// Round 2: add GLOBAL_LOAD_ASYNC_TO_LDS_B128 (ASYNCcnt) for GEMM A-tile
// staging and the attention K-row gather.
// ---------------------------------------------------------------------------

typedef __bf16 bf16_t;
typedef __attribute__((ext_vector_type(16))) __bf16 v16bf;
typedef __attribute__((ext_vector_type(8)))  float  v8f;

union Frag16 {
  v16bf  v;
  bf16_t e[16];
  uint4  q[2];
};

#define ZERO8 {0.f,0.f,0.f,0.f,0.f,0.f,0.f,0.f}

static __device__ inline v8f wmma_bf16(v16bf a, v16bf b, v8f c) {
  // (neg_a, A, neg_b, B, c_mod, C, reuse_a, reuse_b)
  return __builtin_amdgcn_wmma_f32_16x16x32_bf16(false, a, false, b,
                                                 (short)0, c, false, false);
}

// Async memory->LDS copy, 16 bytes, tracked by ASYNCcnt (CDNA5 path).
// LDS byte address = low 32 bits of the generic pointer (HW adds LDS_BASE).
static __device__ inline void async_load_lds_b128(void* lds_dst,
                                                  const void* gsrc) {
  unsigned off = (unsigned)(uintptr_t)lds_dst;
  unsigned long long ga = (unsigned long long)(uintptr_t)gsrc;
  asm volatile("global_load_async_to_lds_b128 %0, %1, off"
               :: "v"(off), "v"(ga) : "memory");
}
static __device__ inline void wait_asynccnt0() {
  asm volatile("s_wait_asynccnt 0x0" ::: "memory");
}

// A fragment: 16x32 bf16 tile, row-major source, leading dim `ld` elements.
// Lane L (wave32): row = L%16, half = L/16; VGPR0..3 hold K=half*8+0..7,
// VGPR4..7 hold K=16+half*8+0..7  => two contiguous 16B loads.
static __device__ inline v16bf frag_a(const bf16_t* base, int ld, int lane) {
  int row = lane & 15, half = lane >> 4;
  const bf16_t* p = base + (size_t)row * ld + half * 8;
  Frag16 f;
  f.q[0] = *(const uint4*)p;
  f.q[1] = *(const uint4*)(p + 16);
  return f.v;
}

// B fragment: 32x16 bf16 tile from TRANSPOSED storage baseT[n][k], ld elems.
// Lane L: col = L%16, half = L/16; VGPR0..3 hold K=half*8+0..7,
// VGPR4..7 hold K=16+half*8+0..7  => two contiguous 16B loads.
static __device__ inline v16bf frag_bT(const bf16_t* baseT, int ld, int lane) {
  int col = lane & 15, half = lane >> 4;
  const bf16_t* p = baseT + (size_t)col * ld + half * 8;
  Frag16 f;
  f.q[0] = *(const uint4*)p;
  f.q[1] = *(const uint4*)(p + 16);
  return f.v;
}

// ---------------------------------------------------------------------------
// 1) elementwise f32 -> bf16 cast
// ---------------------------------------------------------------------------
__global__ __launch_bounds__(256) void cast_bf16_kernel(
    const float* __restrict__ in, bf16_t* __restrict__ out, int n) {
  int i = blockIdx.x * 256 + threadIdx.x;
  if (i < n) out[i] = (bf16_t)in[i];
}

// ---------------------------------------------------------------------------
// 2) QKV projection GEMM: [8192,512]bf16 @ [512,1536]bf16, f32 accum,
//    epilogue scatters to q/k/v buffers of layout [B,H,N,D] (bf16).
//    Column c of Wqkv maps to (h = c/192, d = (c%192)/3, sel = c%3).
// ---------------------------------------------------------------------------
__global__ __launch_bounds__(256) void qkv_gemm_kernel(
    const bf16_t* __restrict__ A, const bf16_t* __restrict__ Bm,
    bf16_t* __restrict__ q, bf16_t* __restrict__ k, bf16_t* __restrict__ v) {
  const int Kd = 512, Nn = 1536;
  __shared__ bf16_t As[64][32];
  __shared__ bf16_t BsT[64][32];   // transposed tile: BsT[n][k]
  int m0 = blockIdx.y * 64, n0 = blockIdx.x * 64;
  int t = threadIdx.x, wave = t >> 5, lane = t & 31;
  int mt = wave >> 1, ntb = (wave & 1) * 2;   // wave owns 2 of 16 C tiles
  v8f acc0 = ZERO8, acc1 = ZERO8;

  for (int k0 = 0; k0 < Kd; k0 += 32) {
    {   // stage A tile 64x32: async memory->LDS (no VGPR round-trip)
      int row = t >> 2, c8 = (t & 3) * 8;
      async_load_lds_b128(&As[row][c8],
                          A + (size_t)(m0 + row) * Kd + k0 + c8);
    }
    {   // stage B tile 32x64, transposed into BsT (needs VGPR transpose)
      int kr = t >> 3, c8 = (t & 7) * 8;
      uint4 d = *(const uint4*)(Bm + (size_t)(k0 + kr) * Nn + n0 + c8);
      const bf16_t* e = (const bf16_t*)&d;
#pragma unroll
      for (int i = 0; i < 8; ++i) BsT[c8 + i][kr] = e[i];
    }
    wait_asynccnt0();
    __syncthreads();
    v16bf a  = frag_a(&As[mt * 16][0], 32, lane);
    v16bf b0 = frag_bT(&BsT[ntb * 16][0], 32, lane);
    v16bf b1 = frag_bT(&BsT[(ntb + 1) * 16][0], 32, lane);
    acc0 = wmma_bf16(a, b0, acc0);
    acc1 = wmma_bf16(a, b1, acc1);
    __syncthreads();
  }

#pragma unroll
  for (int tt = 0; tt < 2; ++tt) {
    v8f acc = tt ? acc1 : acc0;
    int n = n0 + (ntb + tt) * 16 + (lane & 15);
    int h = n / 192, rem = n % 192, d = rem / 3, sel = rem % 3;
    bf16_t* dst = (sel == 0) ? q : (sel == 1) ? k : v;
#pragma unroll
    for (int r = 0; r < 8; ++r) {
      int m = m0 + mt * 16 + (lane >> 4) * 8 + r;
      int b = m >> 11, np = m & 2047;   // N = 2048
      dst[(((size_t)b * 8 + h) * 2048 + np) * 64 + d] = (bf16_t)acc[r];
    }
  }
}

// ---------------------------------------------------------------------------
// 3) Top-K (K=256 smallest of 2048) per (b,h,nw); ascending-index output.
//    Radix binary search on order-preserving uint key, then stable scans.
// ---------------------------------------------------------------------------
static __device__ inline int block_scan_excl(int val, int t, int* buf,
                                             int* total) {
  buf[t] = val;
  __syncthreads();
  for (int off = 1; off < 256; off <<= 1) {
    int vv = (t >= off) ? buf[t - off] : 0;
    __syncthreads();
    buf[t] += vv;
    __syncthreads();
  }
  int incl = buf[t];
  *total = buf[255];
  __syncthreads();
  return incl - val;
}

__global__ __launch_bounds__(256) void topk_kernel(
    const float* __restrict__ pareto, const unsigned char* __restrict__ mask,
    int* __restrict__ idx_out) {
  __shared__ unsigned int skey[2048];
  __shared__ unsigned char sel[2048];
  __shared__ int scan_buf[256];
  __shared__ int cnt_s;
  int wg = blockIdx.x;                       // wg = (b*8 + h)*32 + nw
  int b = wg >> 8, nw = wg & 31;
  int t = threadIdx.x;
  const float* pbase = pareto + (size_t)wg * 2048;

#pragma unroll
  for (int e = 0; e < 8; ++e) {
    int j = t * 8 + e;
    float g = fabsf((float)(nw - (j >> 6)));
    float f = g - pbase[j];
    if (mask[b * 2048 + j]) f = 3.4028235e38f;
    unsigned int u = __float_as_uint(f);
    u = (u & 0x80000000u) ? ~u : (u | 0x80000000u);  // order-preserving
    skey[j] = u;
  }
  __syncthreads();

  // largest prefix with count(key < prefix) < 256  =>  prefix == Kth smallest
  unsigned int prefix = 0;
  for (int bit = 31; bit >= 0; --bit) {
    unsigned int cand = prefix | (1u << bit);
    if (t == 0) cnt_s = 0;
    __syncthreads();
    int local = 0;
#pragma unroll
    for (int e = 0; e < 8; ++e) local += (skey[t * 8 + e] < cand) ? 1 : 0;
    atomicAdd(&cnt_s, local);
    __syncthreads();
    int total = cnt_s;
    __syncthreads();
    if (total < 256) prefix = cand;
  }
  const unsigned int T = prefix;

  // rank the == T elements so ties are broken by ascending index
  int ceq = 0;
#pragma unroll
  for (int e = 0; e < 8; ++e) ceq += (skey[t * 8 + e] == T) ? 1 : 0;
  int totLess;
  {
    int cless = 0;
#pragma unroll
    for (int e = 0; e < 8; ++e) cless += (skey[t * 8 + e] < T) ? 1 : 0;
    (void)block_scan_excl(cless, t, scan_buf, &totLess);
  }
  int totEq;
  int eqExcl = block_scan_excl(ceq, t, scan_buf, &totEq);
  int budget = 256 - totLess;
  int er = eqExcl;
#pragma unroll
  for (int e = 0; e < 8; ++e) {
    int j = t * 8 + e;
    unsigned int u = skey[j];
    int s = (u < T) ? 1 : 0;
    if (u == T) { s = (er < budget) ? 1 : 0; er++; }
    sel[j] = (unsigned char)s;
  }
  __syncthreads();

  // final stable compaction in pure index order
  int csel = 0;
#pragma unroll
  for (int e = 0; e < 8; ++e) csel += sel[t * 8 + e];
  int totSel;
  int pos = block_scan_excl(csel, t, scan_buf, &totSel);
  int* out = idx_out + (size_t)wg * 256;
#pragma unroll
  for (int e = 0; e < 8; ++e) {
    int j = t * 8 + e;
    if (sel[j]) { if (pos < 256) out[pos] = j; pos++; }
  }
}

// ---------------------------------------------------------------------------
// 4) Fused attention per (b,h,nw): gather K/V, QK^T (WMMA), +scale+pos_bias,
//    mask, softmax, attn@V (WMMA). Dynamic LDS = 130 KB (<< 320 KB/WGP).
// ---------------------------------------------------------------------------
#define ATTN_LDS_BYTES (2048 + 32768 + 32768 + 65536)

__global__ __launch_bounds__(256) void attn_kernel(
    const bf16_t* __restrict__ qb, const bf16_t* __restrict__ kb,
    const bf16_t* __restrict__ vb, const int* __restrict__ idx,
    const float* __restrict__ pos_bias, const unsigned char* __restrict__ mask,
    bf16_t* __restrict__ ao) {
  extern __shared__ char smem[];
  int*    idx_s   = (int*)smem;                              // 256 ints
  int*    kmask_s = (int*)(smem + 1024);                     // 256 ints
  bf16_t* ksel    = (bf16_t*)(smem + 2048);                  // [256][64]
  bf16_t* vT      = (bf16_t*)(smem + 2048 + 32768);          // [64][256]
  float*  sc      = (float*)(smem + 2048 + 65536);           // [64][256]

  const int wg = blockIdx.x;
  const int b = wg >> 8, h = (wg >> 5) & 7, nw = wg & 31;
  const int t = threadIdx.x, wave = t >> 5, lane = t & 31;
  const size_t bh = (size_t)b * 8 + h;
  const float SCALE = 0.125f;   // 64^-0.5

  // ---- phase 1: indices, kv mask, prefetch pos_bias gather targets -------
  int id = idx[(size_t)wg * 256 + t] & 2047;
  idx_s[t] = id;
  kmask_s[t] = mask[b * 2048 + id] ? 1 : 0;
  const float* pb0 = pos_bias + ((size_t)h * 2048 + (size_t)nw * 64) * 2048;
#pragma unroll
  for (int mm = 0; mm < 64; mm += 8)
    __builtin_prefetch(pb0 + (size_t)mm * 2048 + id, 0, 0);
  __syncthreads();

  // ---- phase 2: gather K rows async into LDS; V rows transposed ----------
  {
    const bf16_t* krow = kb + (bh * 2048 + idx_s[t]) * 64;
    bf16_t* dstrow = ksel + (size_t)t * 64;
#pragma unroll
    for (int i = 0; i < 8; ++i)        // 8 x 16B async memory->LDS copies
      async_load_lds_b128(dstrow + i * 8, krow + i * 8);
    const bf16_t* vrow = vb + (bh * 2048 + idx_s[t]) * 64;
#pragma unroll
    for (int d = 0; d < 64; ++d) vT[(size_t)d * 256 + t] = vrow[d];
  }
  wait_asynccnt0();
  __syncthreads();

  // ---- phase 3: scores = Q @ Ksel^T (M=64, N=256, Kdim=64) ---------------
  {
    int mt = wave >> 1, nh = wave & 1;
    const bf16_t* qbase = qb + (bh * 2048 + (size_t)nw * 64) * 64;
    v16bf aq[2];
#pragma unroll
    for (int ks = 0; ks < 2; ++ks)
      aq[ks] = frag_a(qbase + (size_t)(mt * 16) * 64 + ks * 32, 64, lane);
#pragma unroll
    for (int j = 0; j < 8; ++j) {
      int nt = nh * 8 + j;
      v8f acc = ZERO8;
#pragma unroll
      for (int ks = 0; ks < 2; ++ks) {
        v16bf bf = frag_bT(ksel + (size_t)(nt * 16) * 64 + ks * 32, 64, lane);
        acc = wmma_bf16(aq[ks], bf, acc);
      }
      int n = nt * 16 + (lane & 15);
      int kidx = idx_s[n];
      int kvm = kmask_s[n];
#pragma unroll
      for (int r = 0; r < 8; ++r) {
        int m = mt * 16 + (lane >> 4) * 8 + r;
        int qm = mask[b * 2048 + nw * 64 + m] ? 1 : 0;
        float val = acc[r] * SCALE + pb0[(size_t)m * 2048 + kidx];
        if (qm & kvm) val = -3.4028235e38f;
        sc[(size_t)m * 256 + n] = val;
      }
    }
  }
  __syncthreads();

  // ---- phase 4: row softmax (wave32 shuffle reductions), in place --------
#pragma unroll
  for (int rr = 0; rr < 8; ++rr) {
    int m = wave * 8 + rr;
    float* row = sc + (size_t)m * 256;
    float vals[8];
    float mx = -INFINITY;
#pragma unroll
    for (int i = 0; i < 8; ++i) {
      vals[i] = row[lane + i * 32];
      mx = fmaxf(mx, vals[i]);
    }
#pragma unroll
    for (int o = 16; o > 0; o >>= 1) mx = fmaxf(mx, __shfl_xor(mx, o, 32));
    float s = 0.f;
#pragma unroll
    for (int i = 0; i < 8; ++i) { vals[i] = __expf(vals[i] - mx); s += vals[i]; }
#pragma unroll
    for (int o = 16; o > 0; o >>= 1) s += __shfl_xor(s, o, 32);
    float inv = 1.0f / s;
#pragma unroll
    for (int i = 0; i < 8; ++i) row[lane + i * 32] = vals[i] * inv;
  }
  __syncthreads();

  // ---- phase 5: out = attn @ Vsel (M=64, N=64, Kdim=256) -----------------
#pragma unroll
  for (int tt = 0; tt < 2; ++tt) {
    int tile = wave * 2 + tt;
    int mt2 = tile >> 2, nt2 = tile & 3;
    v8f acc = ZERO8;
#pragma unroll
    for (int ks = 0; ks < 8; ++ks) {
      // A fragment: f32 attn from LDS, converted to bf16 in registers
      int arow = mt2 * 16 + (lane & 15);
      const float* p0 = sc + (size_t)arow * 256 + ks * 32 + (lane >> 4) * 8;
      const float* p1 = p0 + 16;
      Frag16 f;
#pragma unroll
      for (int i = 0; i < 8; ++i) f.e[i] = (bf16_t)p0[i];
#pragma unroll
      for (int i = 0; i < 8; ++i) f.e[8 + i] = (bf16_t)p1[i];
      v16bf bf = frag_bT(vT + (size_t)(nt2 * 16) * 256 + ks * 32, 256, lane);
      acc = wmma_bf16(f.v, bf, acc);
    }
    int n = nt2 * 16 + (lane & 15);   // d index within head
#pragma unroll
    for (int r = 0; r < 8; ++r) {
      int m = mt2 * 16 + (lane >> 4) * 8 + r;
      ao[((size_t)b * 2048 + nw * 64 + m) * 512 + h * 64 + n] = (bf16_t)acc[r];
    }
  }
}

// ---------------------------------------------------------------------------
// 5) Output projection GEMM: [8192,512]bf16 @ [512,512]bf16 -> f32 d_out
// ---------------------------------------------------------------------------
__global__ __launch_bounds__(256) void out_gemm_kernel(
    const bf16_t* __restrict__ A, const bf16_t* __restrict__ Bm,
    float* __restrict__ out) {
  const int Kd = 512, Nn = 512;
  __shared__ bf16_t As[64][32];
  __shared__ bf16_t BsT[64][32];
  int m0 = blockIdx.y * 64, n0 = blockIdx.x * 64;
  int t = threadIdx.x, wave = t >> 5, lane = t & 31;
  int mt = wave >> 1, ntb = (wave & 1) * 2;
  v8f acc0 = ZERO8, acc1 = ZERO8;

  for (int k0 = 0; k0 < Kd; k0 += 32) {
    {   // async memory->LDS A-tile staging
      int row = t >> 2, c8 = (t & 3) * 8;
      async_load_lds_b128(&As[row][c8],
                          A + (size_t)(m0 + row) * Kd + k0 + c8);
    }
    {
      int kr = t >> 3, c8 = (t & 7) * 8;
      uint4 d = *(const uint4*)(Bm + (size_t)(k0 + kr) * Nn + n0 + c8);
      const bf16_t* e = (const bf16_t*)&d;
#pragma unroll
      for (int i = 0; i < 8; ++i) BsT[c8 + i][kr] = e[i];
    }
    wait_asynccnt0();
    __syncthreads();
    v16bf a  = frag_a(&As[mt * 16][0], 32, lane);
    v16bf b0 = frag_bT(&BsT[ntb * 16][0], 32, lane);
    v16bf b1 = frag_bT(&BsT[(ntb + 1) * 16][0], 32, lane);
    acc0 = wmma_bf16(a, b0, acc0);
    acc1 = wmma_bf16(a, b1, acc1);
    __syncthreads();
  }

#pragma unroll
  for (int tt = 0; tt < 2; ++tt) {
    v8f acc = tt ? acc1 : acc0;
    int n = n0 + (ntb + tt) * 16 + (lane & 15);
#pragma unroll
    for (int r = 0; r < 8; ++r) {
      int m = m0 + mt * 16 + (lane >> 4) * 8 + r;
      out[(size_t)m * 512 + n] = acc[r];
    }
  }
}

// ---------------------------------------------------------------------------
// launch: inputs in setup_inputs() order: x, mask, pareto, pos_bias, Wqkv, Wout
// ---------------------------------------------------------------------------
extern "C" void kernel_launch(void* const* d_in, const int* in_sizes, int n_in,
                              void* d_out, int out_size, void* d_ws,
                              size_t ws_size, hipStream_t stream) {
  (void)in_sizes; (void)n_in; (void)out_size; (void)ws_size;
  const float*         x        = (const float*)d_in[0];
  const unsigned char* mask     = (const unsigned char*)d_in[1];
  const float*         pareto   = (const float*)d_in[2];
  const float*         pos_bias = (const float*)d_in[3];
  const float*         Wqkv     = (const float*)d_in[4];
  const float*         Wout     = (const float*)d_in[5];
  float*               out      = (float*)d_out;

  // workspace carve-up (~45 MB total)
  char* ws = (char*)d_ws;
  size_t off = 0;
  auto carve = [&](size_t bytes) -> void* {
    off = (off + 255) & ~(size_t)255;
    void* p = ws + off;
    off += bytes;
    return p;
  };
  bf16_t* xb    = (bf16_t*)carve((size_t)8192 * 512 * 2);
  bf16_t* wqkvb = (bf16_t*)carve((size_t)512 * 1536 * 2);
  bf16_t* woutb = (bf16_t*)carve((size_t)512 * 512 * 2);
  bf16_t* qbuf  = (bf16_t*)carve((size_t)4 * 8 * 2048 * 64 * 2);
  bf16_t* kbuf  = (bf16_t*)carve((size_t)4 * 8 * 2048 * 64 * 2);
  bf16_t* vbuf  = (bf16_t*)carve((size_t)4 * 8 * 2048 * 64 * 2);
  int*    idxb  = (int*)carve((size_t)1024 * 256 * 4);
  bf16_t* aob   = (bf16_t*)carve((size_t)8192 * 512 * 2);

  // 1) casts to bf16
  {
    int n = 8192 * 512;
    cast_bf16_kernel<<<(n + 255) / 256, 256, 0, stream>>>(x, xb, n);
  }
  {
    int n = 512 * 1536;
    cast_bf16_kernel<<<(n + 255) / 256, 256, 0, stream>>>(Wqkv, wqkvb, n);
  }
  {
    int n = 512 * 512;
    cast_bf16_kernel<<<(n + 255) / 256, 256, 0, stream>>>(Wout, woutb, n);
  }

  // 2) QKV projection (WMMA) with scatter to [B,H,N,D]
  qkv_gemm_kernel<<<dim3(1536 / 64, 8192 / 64), 256, 0, stream>>>(
      xb, wqkvb, qbuf, kbuf, vbuf);

  // 3) top-K key selection per (b,h,window)
  topk_kernel<<<1024, 256, 0, stream>>>(pareto, mask, idxb);

  // 4) fused gather + QK^T + bias + softmax + attn@V (WMMA), 130KB LDS
  (void)hipFuncSetAttribute((const void*)attn_kernel,
                            hipFuncAttributeMaxDynamicSharedMemorySize,
                            ATTN_LDS_BYTES);
  attn_kernel<<<1024, 256, ATTN_LDS_BYTES, stream>>>(
      qbuf, kbuf, vbuf, idxb, pos_bias, mask, aob);

  // 5) output projection (WMMA) -> f32 d_out
  out_gemm_kernel<<<dim3(512 / 64, 8192 / 64), 256, 0, stream>>>(
      aob, woutb, out);
}
